// InverseRecurrentLayer_56212531970294
// MI455X (gfx1250) — matrix operand order repaired and checked by reading, not compiled, not run on hardware
//
#include <hip/hip_runtime.h>
#include <hip/hip_bf16.h>

// ---------------------------------------------------------------------------
// InverseRecurrentLayer on MI455X (gfx1250, wave32, WMMA + TDM)
//
//   states[t] = tanh( x_t @ R + states[t-1] @ Wt + bias ),  Wt = W or W^{-1}=W^T
//
// Persistent kernel, 32 WGs x 256 threads. Per step each wave computes one
// 16x16 tile of the [64 x 1024] state with v_wmma_f32_16x16x32_f16 over
// K = 1024 (recurrence) + 512 (input projection, fused). Weight strips live
// in LDS (one-time load). The f16 state ping-pong buffer is staged into LDS
// each step by the Tensor Data Mover (tensor_load_to_lds, TENSORcnt) when the
// builtin is available, else by a manual vector copy. Steps are separated by
// a software grid barrier (re-initialized every launch -> graph-replay safe).
// ---------------------------------------------------------------------------

#define UNITS   1024
#define FT      512
#define TSTEPS  512
#define BATCH   64
#define NWG     32
#define NTHREADS 256
#define STRIP   32      // output columns per workgroup

typedef _Float16 half8  __attribute__((ext_vector_type(8)));
typedef _Float16 half16 __attribute__((ext_vector_type(16)));
typedef float    float8 __attribute__((ext_vector_type(8)));
typedef unsigned v4u    __attribute__((ext_vector_type(4)));
typedef int      v4i    __attribute__((ext_vector_type(4)));
typedef int      v8i    __attribute__((ext_vector_type(8)));

#if __has_builtin(__builtin_amdgcn_tensor_load_to_lds) && \
    __has_builtin(__builtin_amdgcn_s_wait_tensorcnt)
#define USE_TDM 1
#else
#define USE_TDM 0
#endif

__device__ __forceinline__ half16 h16cat(half8 lo, half8 hi) {
  return __builtin_shufflevector(lo, hi, 0,1,2,3,4,5,6,7,8,9,10,11,12,13,14,15);
}

// LDS byte offset of a generic pointer to __shared__: the LDS aperture keeps
// the offset in addr[31:0] (ISA 10.2 per-aperture address calculation).
__device__ __forceinline__ unsigned lds_addr32(const void* p) {
  return (unsigned)(unsigned long long)p;
}

// Sense-reversal grid barrier in global memory. bar[0]=arrive count, bar[1]=gen.
__device__ __forceinline__ void grid_sync(unsigned* bar) {
  __syncthreads();
  if (threadIdx.x == 0) {
    __threadfence();
    volatile unsigned* vb = (volatile unsigned*)bar;
    unsigned gen = vb[1];
    unsigned prev = atomicAdd(&bar[0], 1u);
    if (prev == gridDim.x - 1) {
      vb[0] = 0u;
      __threadfence();
      atomicAdd(&bar[1], 1u);
    } else {
      while (vb[1] == gen) { __builtin_amdgcn_s_sleep(2); }
    }
    __threadfence();
  }
  __syncthreads();
}

// ---------------- one-time prep kernels (fully parallel) -------------------

// WT_h[n][k]    = (f16) W[k][n]      (B-operand layout for Wt = W)
// WinvT_h[n][k] = (f16) W[n][k]      (since W^{-1} = W^T for orthogonal W)
// RT_h[n][f]    = (f16) R[f][n]
__global__ void k_prep_wr(const float* __restrict__ W, const float* __restrict__ R,
                          _Float16* __restrict__ WT_h, _Float16* __restrict__ WinvT_h,
                          _Float16* __restrict__ RT_h) {
  int i = blockIdx.x * blockDim.x + threadIdx.x;
  if (i < UNITS * UNITS) {
    int n = i >> 10, k = i & (UNITS - 1);
    WinvT_h[i] = (_Float16)W[i];                       // W[n][k]
    WT_h[i]    = (_Float16)W[(size_t)k * UNITS + n];   // W[k][n]
  }
  if (i < FT * UNITS) {
    int n = i >> 9, f = i & (FT - 1);
    RT_h[i] = (_Float16)R[(size_t)f * UNITS + n];      // R[f][n]
  }
}

// xH[t][b][f] = (f16) inputs[b][t][f]
__global__ void k_prep_x(const float* __restrict__ inputs, _Float16* __restrict__ xH) {
  size_t i = (size_t)blockIdx.x * blockDim.x + threadIdx.x;
  if (i >= (size_t)TSTEPS * BATCH * FT) return;
  int f = (int)(i & (FT - 1));
  size_t rem = i >> 9;             // t*64 + b
  int b = (int)(rem & (BATCH - 1));
  int t = (int)(rem >> 6);
  xH[i] = (_Float16)inputs[(((size_t)b * TSTEPS + t) << 9) + f];
}

// state buffer 0 <- broadcast x0 ; reset barrier (every launch: deterministic)
__global__ void k_prep_state(const float* __restrict__ x0, _Float16* __restrict__ stateH,
                             unsigned* __restrict__ bar) {
  int i = blockIdx.x * blockDim.x + threadIdx.x;
  if (i < BATCH * UNITS) stateH[i] = (_Float16)x0[i & (UNITS - 1)];
  if (i == 0) { bar[0] = 0u; bar[1] = 0u; }
}

// ---------------- persistent recurrence kernel -----------------------------

__global__ void __launch_bounds__(NTHREADS, 1)
k_rnn_persistent(const _Float16* __restrict__ xH,
                 const _Float16* __restrict__ WT_h,
                 const _Float16* __restrict__ WinvT_h,
                 const _Float16* __restrict__ RT_h,
                 const float*    __restrict__ bias,
                 _Float16*       __restrict__ stateH,   // 2 x [64][1024] f16, ping-pong
                 unsigned*       __restrict__ bar,
                 float*          __restrict__ out)      // [T][64][1024] f32
{
  // 288 KB static LDS — CDNA5 WGP allows 320 KB per workgroup.
  __shared__ __align__(16) _Float16 sState[BATCH * UNITS];   // 128 KB
  __shared__ __align__(16) _Float16 sWT[STRIP * UNITS];      //  64 KB
  __shared__ __align__(16) _Float16 sWinvT[STRIP * UNITS];   //  64 KB
  __shared__ __align__(16) _Float16 sRT[STRIP * FT];         //  32 KB

  const int tid   = threadIdx.x;
  const int lane  = tid & 31;
  const int wave  = tid >> 5;         // 0..7
  const int mtile = wave & 3;         // row tile: rows [16*mtile, +16)
  const int nsub  = wave >> 2;        // 0..1: 16-column subtile within strip
  const int c0wg  = blockIdx.x * STRIP;
  const int cbl   = nsub * 16;

  // One-time LDS preload of this WG's weight column strips ([n][k] layout).
  for (int i = tid; i < STRIP * UNITS; i += NTHREADS) {
    int nl = i >> 10, k = i & (UNITS - 1);
    sWT[i]    = WT_h[(size_t)(c0wg + nl) * UNITS + k];
    sWinvT[i] = WinvT_h[(size_t)(c0wg + nl) * UNITS + k];
  }
  for (int i = tid; i < STRIP * FT; i += NTHREADS) {
    int nl = i >> 9, f = i & (FT - 1);
    sRT[i] = RT_h[(size_t)(c0wg + nl) * FT + f];
  }

  // Per-lane fragment geometry (wave32 WMMA layouts, ISA 7.12.2):
  //   A: lane holds row M=lane&15, K chunks [kb,kb+8) and [kb+16,kb+24), kb=8*(lane>=16)
  //   B: lane holds col N=lane&15, K chunks [kb,kb+8) and [kb+16,kb+24), kb=8*(lane>=16)
  //   C/D: VGPR r -> (M = r + 8*(lane>=16), N = lane&15)
  const int colL  = lane & 15;
  const int colg  = c0wg + cbl + colL;
  const float biasv = bias[colg];
  const int rb    = mtile * 16;
  const int rowA  = rb + colL;
  const int kaoff = (lane >> 4) << 3;   // 0 or 8
  const int kboff = (lane >> 4) << 4;   // 0 or 16

  const _Float16* aState = sState + (size_t)rowA * UNITS;
  const _Float16* bW     = sWT    + (size_t)(cbl + colL) * UNITS;
  const _Float16* bWinv  = sWinvT + (size_t)(cbl + colL) * UNITS;
  const _Float16* bR     = sRT    + (size_t)(cbl + colL) * FT;

#if USE_TDM
  // Tensor-DMA descriptor constants for the 128 KB state tile:
  // view it as 32768 dwords = 4 rows x 8192 dwords (data_size code 2 = 4B).
  const unsigned sStateLds = lds_addr32(sState);
#endif

  for (int t = 0; t < TSTEPS; ++t) {
    // ---- Stage state(t) (f16, 128 KB) from global ping-pong buffer into LDS.
    const _Float16* stSrc = stateH + (size_t)(t & 1) * (BATCH * UNITS);
#if USE_TDM
    if (wave == 0) {
      unsigned long long ga = (unsigned long long)stSrc;
      v4u g0;
      g0.x = 1u;                                   // count=1, user descriptor
      g0.y = sStateLds;                            // lds_addr (bytes)
      g0.z = (unsigned)(ga & 0xffffffffu);         // global_addr[31:0]
      g0.w = (unsigned)((ga >> 32) & 0x01ffffffu)  // global_addr[56:32]
             | 0x80000000u;                        // type=2 ("image")
      v8i g1;
      g1[0] = (int)(2u << 16);                     // data_size = 4B
      g1[1] = (int)(8192u << 16);                  // tensor_dim0 = 8192 (lo16)
      g1[2] = (int)(4u << 16);                     // tensor_dim0 hi=0, tensor_dim1 = 4
      g1[3] = (int)(8192u << 16);                  // tensor_dim1 hi=0, tile_dim0 = 8192
      g1[4] = 4;                                   // tile_dim1 = 4, tile_dim2 = 0
      g1[5] = 8192;                                // tensor_dim0_stride = 8192
      g1[6] = 0;
      g1[7] = 0;
      v4i gz = {0, 0, 0, 0};
#if __clang_major__ >= 23
      v8i gz8 = {0, 0, 0, 0, 0, 0, 0, 0};
      __builtin_amdgcn_tensor_load_to_lds(g0, g1, gz, gz, gz8, 0);
#else
      __builtin_amdgcn_tensor_load_to_lds(g0, g1, gz, gz, 0);
#endif
      __builtin_amdgcn_s_wait_tensorcnt(0);
    }
#else
    {
      const half8* src = (const half8*)stSrc;
      half8* dst = (half8*)sState;
      #pragma unroll 4
      for (int i = tid; i < (BATCH * UNITS) / 8; i += NTHREADS) dst[i] = src[i];
    }
#endif
    __syncthreads();

    // step t uses W iff (t/64)%2==1, else W^{-1}
    const _Float16* bMat = ((t >> 6) & 1) ? bW : bWinv;
    float8 acc = {0.f, 0.f, 0.f, 0.f, 0.f, 0.f, 0.f, 0.f};

    // state @ Wt : K = 1024
    #pragma unroll 4
    for (int k0 = 0; k0 < UNITS; k0 += 32) {
      half16 a = h16cat(*(const half8*)(aState + k0 + kaoff),
                        *(const half8*)(aState + k0 + kaoff + 16));
      half16 b = h16cat(*(const half8*)(bMat + k0 + kaoff),
                        *(const half8*)(bMat + k0 + kaoff + 16));
      acc = __builtin_amdgcn_wmma_f32_16x16x32_f16(false, a, false, b,
                                                   (short)0, acc, false, false);
    }

    // x_t @ R : K = 512 (A straight from L2-resident f16 inputs);
    // prefetch next step's x rows into cache while the math pipes are busy.
    const _Float16* aX = xH + ((size_t)t * BATCH + rowA) * FT;
    __builtin_prefetch(aX + (size_t)BATCH * FT, 0, 1);
    #pragma unroll 4
    for (int k0 = 0; k0 < FT; k0 += 32) {
      half16 a = h16cat(*(const half8*)(aX + k0 + kaoff),
                        *(const half8*)(aX + k0 + kaoff + 16));
      half16 b = h16cat(*(const half8*)(bR + k0 + kaoff),
                        *(const half8*)(bR + k0 + kaoff + 16));
      acc = __builtin_amdgcn_wmma_f32_16x16x32_f16(false, a, false, b,
                                                   (short)0, acc, false, false);
    }

    // Epilogue: tanh, write f32 output + f16 next-state.
    float*    outT = out    + (size_t)t * (BATCH * UNITS);
    _Float16* stN  = stateH + (size_t)((t + 1) & 1) * (BATCH * UNITS);
    const int rbase = (lane < 16) ? 0 : 8;
    #pragma unroll
    for (int r = 0; r < 8; ++r) {
      const int rowg = rb + rbase + r;
      float v = tanhf(acc[r] + biasv);
      outT[(size_t)rowg * UNITS + colg] = v;
      stN[(size_t)rowg * UNITS + colg] = (_Float16)v;
    }

    grid_sync(bar);   // publish state(t+1) to all WGs
  }
}

// ---------------------------------------------------------------------------

extern "C" void kernel_launch(void* const* d_in, const int* in_sizes, int n_in,
                              void* d_out, int out_size, void* d_ws, size_t ws_size,
                              hipStream_t stream) {
  const float* inputs = (const float*)d_in[0];   // [64][512][512]
  const float* R      = (const float*)d_in[1];   // [512][1024]
  const float* W      = (const float*)d_in[2];   // [1024][1024]
  const float* bias   = (const float*)d_in[3];   // [1024]
  const float* x0     = (const float*)d_in[4];   // [1024]
  float* out = (float*)d_out;                    // [512][64][1024]

  // Workspace layout (~37.3 MB total)
  char* ws = (char*)d_ws;
  size_t off = 0;
  unsigned* bar = (unsigned*)(ws + off);          off += 256;
  _Float16* WT_h    = (_Float16*)(ws + off);      off += (size_t)UNITS * UNITS * 2;
  _Float16* WinvT_h = (_Float16*)(ws + off);      off += (size_t)UNITS * UNITS * 2;
  _Float16* RT_h    = (_Float16*)(ws + off);      off += (size_t)FT * UNITS * 2;
  _Float16* xH      = (_Float16*)(ws + off);      off += (size_t)TSTEPS * BATCH * FT * 2;
  _Float16* stateH  = (_Float16*)(ws + off);      off += (size_t)2 * BATCH * UNITS * 2;
  (void)ws_size; (void)in_sizes; (void)n_in; (void)out_size;

  k_prep_wr<<<(UNITS * UNITS + 255) / 256, 256, 0, stream>>>(W, R, WT_h, WinvT_h, RT_h);
  {
    size_t n = (size_t)TSTEPS * BATCH * FT;
    k_prep_x<<<(unsigned)((n + 255) / 256), 256, 0, stream>>>(inputs, xH);
  }
  k_prep_state<<<(BATCH * UNITS + 255) / 256, 256, 0, stream>>>(x0, stateH, bar);

  k_rnn_persistent<<<NWG, NTHREADS, 0, stream>>>(xH, WT_h, WinvT_h, RT_h,
                                                 bias, stateH, bar, out);
}
